// VulnDetector_71133248357084
// MI455X (gfx1250) — matrix operand order repaired and checked by reading, not compiled, not run on hardware
//
#include <hip/hip_runtime.h>
#include <hip/hip_bf16.h>

#define BATCH   64
#define NNODES  50000
#define NEDGES  300000
#define NFEAT   128
#define GDIM    256
#define HDIM    512
#define NREL    3
#define NLAYERS 2

typedef __attribute__((ext_vector_type(16))) __bf16 v16bf;
typedef __attribute__((ext_vector_type(8)))  float  v8f;

// ---------------------------------------------------------------------------
// Pack an fp32 [K, Ncols] weight matrix into bf16 WMMA B-fragment order:
// element index = (kt*ntilesN + nt)*512 + lane*16 + e
//   K = kt*32 + 8*(lane>>4) + (e&7) + (e>=8 ? 16 : 0) ;  N = nt*16 + (lane&15)
// ---------------------------------------------------------------------------
__global__ void pack_w(const float* __restrict__ W, __bf16* __restrict__ P,
                       int K, int Ncols) {
    int idx = blockIdx.x * blockDim.x + threadIdx.x;
    if (idx >= K * Ncols) return;
    int e    = idx & 15;
    int lane = (idx >> 4) & 31;
    int t    = idx >> 9;
    int ntN  = Ncols >> 4;
    int kt   = t / ntN;
    int nt   = t - kt * ntN;
    int k = kt * 32 + ((lane >> 4) << 3) + (e & 7) + ((e & 8) ? 16 : 0);
    int n = nt * 16 + (lane & 15);
    P[idx] = (__bf16)W[(size_t)k * Ncols + n];
}

// ---------------------------------------------------------------------------
// Pack an fp32 [M, K] activation matrix into bf16 WMMA A-fragment order:
// element index = (rt*KT + kt)*512 + lane*16 + e      (KT = K/32)
//   row = rt*16 + (lane&15) ;  k = kt*32 + 8*(lane>>4) + (e&7) + (e>=8 ? 16 : 0)
// ---------------------------------------------------------------------------
__global__ void pack_a(const float* __restrict__ X, __bf16* __restrict__ P,
                       int M, int K) {
    int idx = blockIdx.x * blockDim.x + threadIdx.x;
    if (idx >= M * K) return;
    int e    = idx & 15;
    int lane = (idx >> 4) & 31;
    int t    = idx >> 9;
    int KT   = K >> 5;
    int kt   = t % KT;
    int rt   = t / KT;
    int row  = rt * 16 + (lane & 15);
    int k    = kt * 32 + ((lane >> 4) << 3) + (e & 7) + ((e & 8) ? 16 : 0);
    P[idx] = (__bf16)X[(size_t)row * K + k];
}

// ---------------------------------------------------------------------------
// x = relu(acc); also emit bf16 A-fragment-packed copy for the next layer.
// One thread per element; K == GDIM == 256.
// ---------------------------------------------------------------------------
__global__ void relu_pack(const float* __restrict__ acc, float* __restrict__ x,
                          __bf16* __restrict__ xpk) {
    size_t i = (size_t)blockIdx.x * blockDim.x + threadIdx.x;
    if (i >= (size_t)NNODES * GDIM) return;
    int row = (int)(i >> 8);
    int k   = (int)(i & 255);
    float v = fmaxf(acc[i], 0.0f);
    x[i] = v;
    int rt = row >> 4, m_ = row & 15;
    int kt = k >> 5,  kk = k & 31;
    int h  = (kk >> 3) & 1;
    int e  = (kk & 7) + ((kk & 16) ? 8 : 0);
    int L  = m_ + (h << 4);
    xpk[(((size_t)rt * (GDIM >> 5) + kt) * 32 + L) * 16 + e] = (__bf16)v;
}

// ---------------------------------------------------------------------------
// C[M, Ncols] = Apk(bf16, A-fragment packed) @ Ppacked + bias (optional)
// Each wave computes a 16x64 strip (4 accumulators, one shared A fragment).
// grid = (ceil(M/16 / 4), Ncols/64), block = 128 (4 waves, distinct row tiles)
// Inner loop: 1 packed-A v16bf load + 4 B v16bf loads + 4 WMMAs, no cvt.
// ---------------------------------------------------------------------------
__global__ __launch_bounds__(128) void gemm_pk_wmma(
    const __bf16* __restrict__ Apk, const __bf16* __restrict__ P,
    const float* __restrict__ bias, float* __restrict__ C,
    int M, int K, int Ncols) {
    const int lane    = threadIdx.x & 31;
    const int wave    = threadIdx.x >> 5;
    const int rowTile = (blockIdx.x << 2) + wave;
    const int row0    = rowTile << 4;
    if (row0 >= M) return;                       // uniform per-wave exit
    const int colTile0 = blockIdx.y << 2;        // 4 consecutive 16-col tiles
    const int m        = lane & 15;
    const int h        = lane >> 4;
    const int ntilesN  = Ncols >> 4;
    const int KT       = K >> 5;

    v8f acc0 = {}, acc1 = {}, acc2 = {}, acc3 = {};
    const __bf16* pa    = Apk + (((size_t)rowTile * KT) * 32 + lane) * 16;
    const __bf16* pb    = P + ((size_t)colTile0 << 9) + ((size_t)lane << 4);
    const size_t  pstep = (size_t)ntilesN << 9;

    for (int kt = 0; kt < KT; ++kt) {
        const v16bf a  = *(const v16bf*)pa;
        const v16bf b0 = *(const v16bf*)(pb);
        const v16bf b1 = *(const v16bf*)(pb + 512);
        const v16bf b2 = *(const v16bf*)(pb + 1024);
        const v16bf b3 = *(const v16bf*)(pb + 1536);
        acc0 = __builtin_amdgcn_wmma_f32_16x16x32_bf16(false, a, false, b0, (short)0, acc0, false, false);
        acc1 = __builtin_amdgcn_wmma_f32_16x16x32_bf16(false, a, false, b1, (short)0, acc1, false, false);
        acc2 = __builtin_amdgcn_wmma_f32_16x16x32_bf16(false, a, false, b2, (short)0, acc2, false, false);
        acc3 = __builtin_amdgcn_wmma_f32_16x16x32_bf16(false, a, false, b3, (short)0, acc3, false, false);
        pa += 512;
        pb += pstep;
    }

    float* cbase = C + (size_t)(row0 + (h << 3)) * Ncols + m;
#pragma unroll
    for (int t = 0; t < 4; ++t) {
        const int col0 = (colTile0 + t) << 4;
        const float bc = bias ? bias[col0 + m] : 0.0f;
        const v8f a = (t == 0) ? acc0 : (t == 1) ? acc1 : (t == 2) ? acc2 : acc3;
        float* crow = cbase + col0;
#pragma unroll
        for (int r = 0; r < 8; ++r)
            crow[(size_t)r * Ncols] = a[r] + bc;
    }
}

// ---------------------------------------------------------------------------
// For edges of relation r: out[dst] += xr[src]  (per-channel f32 atomics)
// ---------------------------------------------------------------------------
__global__ void scatter_rel(const int* __restrict__ ei, const int* __restrict__ et,
                            const float* __restrict__ xr, float* __restrict__ out,
                            int r) {
    int e = blockIdx.x;
    if (et[e] != r) return;
    int src = ei[e];
    int dst = ei[NEDGES + e];
    int c = threadIdx.x;
    atomicAdd(&out[(size_t)dst * GDIM + c], xr[(size_t)src * GDIM + c]);
}

__global__ void zero_f(float* __restrict__ p, int n) {
    int i = blockIdx.x * blockDim.x + threadIdx.x;
    if (i < n) p[i] = 0.0f;
}

__global__ void pool_kernel(const float* __restrict__ x, const int* __restrict__ bv,
                            float* __restrict__ hg, float* __restrict__ cnt) {
    int nidx = blockIdx.x;
    int c    = threadIdx.x;
    int b    = bv[nidx];
    atomicAdd(&hg[(size_t)b * GDIM + c], x[(size_t)nidx * GDIM + c]);
    if (c == 0) atomicAdd(&cnt[b], 1.0f);
}

__global__ void linear_kernel(const float* __restrict__ X, const float* __restrict__ W,
                              const float* __restrict__ bias, float* __restrict__ Y,
                              int M, int K, int Ncols, int do_relu) {
    int idx = blockIdx.x * blockDim.x + threadIdx.x;
    if (idx >= M * Ncols) return;
    int b = idx / Ncols, j = idx - b * Ncols;
    float s = bias[j];
    const float* xr = X + (size_t)b * K;
    for (int k = 0; k < K; ++k) s += xr[k] * W[(size_t)k * Ncols + j];
    if (do_relu) s = fmaxf(s, 0.0f);
    Y[idx] = s;
}

__global__ void meanproj_kernel(const float* __restrict__ hg, const float* __restrict__ cnt,
                                const float* __restrict__ W, const float* __restrict__ bias,
                                float* __restrict__ Y) {
    int idx = blockIdx.x * blockDim.x + threadIdx.x;
    if (idx >= BATCH * HDIM) return;
    int b = idx / HDIM, j = idx - b * HDIM;
    float inv = 1.0f / fmaxf(cnt[b], 1.0f);
    float s = bias[j];
    const float* hr = hg + (size_t)b * GDIM;
    for (int k = 0; k < GDIM; ++k) s += hr[k] * inv * W[(size_t)k * HDIM + j];
    Y[idx] = s;
}

__global__ void gate_fuse_kernel(const float* __restrict__ h, const float* __restrict__ hgp,
                                 const float* __restrict__ gw, const float* __restrict__ gb,
                                 float* __restrict__ f) {
    int idx = blockIdx.x * blockDim.x + threadIdx.x;
    if (idx >= BATCH * HDIM) return;
    int b = idx / HDIM, j = idx - b * HDIM;
    const float* hr  = h   + (size_t)b * HDIM;
    const float* hgr = hgp + (size_t)b * HDIM;
    float s = gb[j];
    for (int k = 0; k < HDIM; ++k) {
        s += hr[k]  * gw[(size_t)k * HDIM + j];
        s += hgr[k] * gw[(size_t)(HDIM + k) * HDIM + j];
    }
    float g = 1.0f / (1.0f + expf(-s));
    f[idx] = g * hr[j] + (1.0f - g) * hgr[j];
}

// ---------------------------------------------------------------------------
extern "C" void kernel_launch(void* const* d_in, const int* in_sizes, int n_in,
                              void* d_out, int out_size, void* d_ws, size_t ws_size,
                              hipStream_t stream) {
    const float* cls     = (const float*)d_in[0];
    const float* nf      = (const float*)d_in[1];
    const int*   ei      = (const int*)d_in[2];
    const int*   et      = (const int*)d_in[3];
    const int*   bv      = (const int*)d_in[4];
    const float* node_w  = (const float*)d_in[5];
    const float* node_b  = (const float*)d_in[6];
    const float* rel_w   = (const float*)d_in[7];
    const float* self_w  = (const float*)d_in[8];
    const float* self_b  = (const float*)d_in[9];
    const float* text_w  = (const float*)d_in[10];
    const float* text_b  = (const float*)d_in[11];
    const float* graph_w = (const float*)d_in[12];
    const float* graph_b = (const float*)d_in[13];
    const float* gate_w  = (const float*)d_in[14];
    const float* gate_b  = (const float*)d_in[15];
    const float* c1_w    = (const float*)d_in[16];
    const float* c1_b    = (const float*)d_in[17];
    const float* c2_w    = (const float*)d_in[18];
    const float* c2_b    = (const float*)d_in[19];
    float* out = (float*)d_out;

    const size_t NG = (size_t)NNODES * GDIM;
    float* x    = (float*)d_ws;
    float* acc  = x   + NG;
    float* xr   = acc + NG;
    float* hg   = xr  + NG;
    float* cnt  = hg   + BATCH * GDIM;
    float* hbuf = cnt  + BATCH;
    float* hgp  = hbuf + BATCH * HDIM;
    float* fbuf = hgp  + BATCH * HDIM;
    float* zbuf = fbuf + BATCH * HDIM;
    __bf16* pnode = (__bf16*)(zbuf + BATCH * (HDIM / 2));
    __bf16* pself = pnode + (size_t)NFEAT * GDIM;
    __bf16* prel  = pself + (size_t)NLAYERS * GDIM * GDIM;
    __bf16* xpk   = prel  + (size_t)NLAYERS * NREL * GDIM * GDIM;  // [N,G] packed
    __bf16* nfpk  = xpk   + NG;                                    // [N,NF] packed

    const size_t GG = (size_t)GDIM * GDIM;

    // GEMM grid: 4 row tiles per block (4 waves), 64 columns per wave
    const unsigned rowBlocks = (NNODES / 16 + 3) / 4;      // 782
    const dim3 ggrid(rowBlocks, GDIM / 64);                // (782, 4)

    // ---- pack all GEMM weights into WMMA B-fragment layout (bf16) ----
    pack_w<<<(NFEAT * GDIM + 255) / 256, 256, 0, stream>>>(node_w, pnode, NFEAT, GDIM);
    for (int l = 0; l < NLAYERS; ++l) {
        pack_w<<<(GG + 255) / 256, 256, 0, stream>>>(self_w + l * GG, pself + l * GG, GDIM, GDIM);
        for (int r = 0; r < NREL; ++r) {
            size_t off = ((size_t)l * NREL + r) * GG;
            pack_w<<<(GG + 255) / 256, 256, 0, stream>>>(rel_w + off, prel + off, GDIM, GDIM);
        }
    }

    // ---- node_proj: x = nf @ node_w + node_b ----
    pack_a<<<(NNODES * NFEAT + 255) / 256, 256, 0, stream>>>(nf, nfpk, NNODES, NFEAT);
    gemm_pk_wmma<<<ggrid, 128, 0, stream>>>(nfpk, pnode, node_b, x, NNODES, NFEAT, GDIM);
    pack_a<<<(unsigned)((NG + 255) / 256), 256, 0, stream>>>(x, xpk, NNODES, GDIM);

    // ---- RGCN layers ----
    for (int l = 0; l < NLAYERS; ++l) {
        // self-loop: acc = x @ self_w[l] + self_b[l]
        gemm_pk_wmma<<<ggrid, 128, 0, stream>>>(
            xpk, pself + l * GG, self_b + (size_t)l * GDIM, acc, NNODES, GDIM, GDIM);
        for (int r = 0; r < NREL; ++r) {
            // per-relation node transform, then gather/scatter over typed edges
            gemm_pk_wmma<<<ggrid, 128, 0, stream>>>(
                xpk, prel + ((size_t)l * NREL + r) * GG, nullptr, xr, NNODES, GDIM, GDIM);
            scatter_rel<<<NEDGES, GDIM, 0, stream>>>(ei, et, xr, acc, r);
        }
        // relu + re-pack activations for the next layer's GEMMs
        relu_pack<<<(unsigned)(NG / 256), 256, 0, stream>>>(acc, x, xpk);
    }

    // ---- mean pooling ----
    zero_f<<<(BATCH * GDIM + BATCH + 255) / 256, 256, 0, stream>>>(hg, BATCH * GDIM + BATCH);
    pool_kernel<<<NNODES, GDIM, 0, stream>>>(x, bv, hg, cnt);

    // ---- heads (tiny, fp32) ----
    linear_kernel<<<(BATCH * HDIM + 255) / 256, 256, 0, stream>>>(
        cls, text_w, text_b, hbuf, BATCH, 768, HDIM, 0);
    meanproj_kernel<<<(BATCH * HDIM + 255) / 256, 256, 0, stream>>>(
        hg, cnt, graph_w, graph_b, hgp);
    gate_fuse_kernel<<<(BATCH * HDIM + 255) / 256, 256, 0, stream>>>(
        hbuf, hgp, gate_w, gate_b, fbuf);
    linear_kernel<<<(BATCH * (HDIM / 2) + 255) / 256, 256, 0, stream>>>(
        fbuf, c1_w, c1_b, zbuf, BATCH, HDIM, HDIM / 2, 1);
    linear_kernel<<<(BATCH * 2 + 255) / 256, 256, 0, stream>>>(
        zbuf, c2_w, c2_b, out, BATCH, HDIM / 2, 2, 0);
}
